// MultiHeadKvtAttention_1683627180143
// MI455X (gfx1250) — matrix operand order repaired and checked by reading, not compiled, not run on hardware
//
#include <hip/hip_runtime.h>

// ---------------- constants ----------------
#define BATCH   8
#define SEQ     1024
#define DIM     768
#define HEADS   12
#define HDIM    64
#define TOPK    100
#define SCALE   0.125f   // 64^-0.5

// ---------------- vector types ----------------
typedef __attribute__((ext_vector_type(16))) __bf16         v16bf;
typedef __attribute__((ext_vector_type(16))) unsigned short v16us;
typedef __attribute__((ext_vector_type(8)))  unsigned short v8us;
typedef __attribute__((ext_vector_type(8)))  float          v8f;

// f32 -> bf16 round-to-nearest-even (bit manipulation, no __bf16 arithmetic)
__device__ __forceinline__ unsigned short f2bf(float f) {
    unsigned int u = __float_as_uint(f);
    unsigned int r = u + 0x7fffu + ((u >> 16) & 1u);
    return (unsigned short)(r >> 16);
}

__device__ __forceinline__ v16bf make_frag(v8us lo, v8us hi) {
    v16us u = __builtin_shufflevector(lo, hi, 0,1,2,3,4,5,6,7,8,9,10,11,12,13,14,15);
    return __builtin_bit_cast(v16bf, u);
}

// A fragment (16x32 bf16, row-major src with leading dim ld):
// lane: m = lane&15, half = lane>>4; K chunks [k0+half*8, +8) and [k0+16+half*8, +8)
__device__ __forceinline__ v16bf load_a_frag(const unsigned short* base, int ld,
                                             int row0, int k0, int lane) {
    int m = lane & 15, half = lane >> 4;
    const unsigned short* p = base + (size_t)(row0 + m) * ld + k0 + half * 8;
    v8us lo = *(const v8us*)(p);
    v8us hi = *(const v8us*)(p + 16);
    return make_frag(lo, hi);
}

// B fragment (32x16 bf16). src layout: [ncol][K] (column's K contiguous), ld = K stride.
// lane: n = lane&15, khalf = lane>>4; K chunks [k0+khalf*8, +8) and [k0+16+khalf*8, +8)
__device__ __forceinline__ v16bf load_b_frag(const unsigned short* base, int ld,
                                             int k0, int n0, int lane) {
    int n = lane & 15, khalf = lane >> 4;
    const unsigned short* p = base + (size_t)(n0 + n) * ld + k0 + khalf * 8;
    v8us lo = *(const v8us*)(p);
    v8us hi = *(const v8us*)(p + 16);
    return make_frag(lo, hi);
}

__device__ __forceinline__ void prefetch_b_frag(const unsigned short* base, int ld,
                                                int k0, int n0, int lane) {
    int n = lane & 15, khalf = lane >> 4;
    __builtin_prefetch(base + (size_t)(n0 + n) * ld + k0 + khalf * 16, 0, 0);
}

__device__ __forceinline__ v8f wmma_bf16(v16bf a, v16bf b, v8f c) {
    return __builtin_amdgcn_wmma_f32_16x16x32_bf16(false, a, false, b,
                                                   (short)0, c, false, false);
}

// ---------------- elementwise packers ----------------
__global__ void cast_bf16_kernel(const float* __restrict__ in,
                                 unsigned short* __restrict__ out, int n) {
    int i = blockIdx.x * 256 + threadIdx.x;
    if (i < n) out[i] = f2bf(in[i]);
}

// in [rows][cols] f32 -> out [cols][rows] bf16
__global__ void transpose_bf16_kernel(const float* __restrict__ in,
                                      unsigned short* __restrict__ out,
                                      int rows, int cols) {
    int i = blockIdx.x * 256 + threadIdx.x;
    if (i >= rows * cols) return;
    int r = i / cols, c = i % cols;
    out[(size_t)c * rows + r] = f2bf(in[i]);
}

// ---------------- QKV projection GEMM ----------------
// xb [8192][768] bf16 (A), wT [2304][768] bf16 (B, K-contig per out col)
// epilogue scatters: q,k -> [B*H][N][64] row-major ; v -> vT [B*H][64][N]
__global__ __launch_bounds__(256)
void qkv_gemm_kernel(const unsigned short* __restrict__ xb,
                     const unsigned short* __restrict__ wT,
                     unsigned short* __restrict__ qb,
                     unsigned short* __restrict__ kb,
                     unsigned short* __restrict__ vT) {
    const int lane = threadIdx.x & 31, w = threadIdx.x >> 5;
    const int row0 = blockIdx.x * 64 + (w & 3) * 16;
    const int col0 = blockIdx.y * 128 + (w >> 2) * 64;

    v8f acc[4];
    #pragma unroll
    for (int t = 0; t < 4; ++t) acc[t] = (v8f){0,0,0,0,0,0,0,0};

    for (int kk = 0; kk < DIM; kk += 32) {
        v16bf a = load_a_frag(xb, DIM, row0, kk, lane);
        if (kk + 32 < DIM) {
            #pragma unroll
            for (int t = 0; t < 4; ++t)
                prefetch_b_frag(wT, DIM, kk + 32, col0 + t * 16, lane);
        }
        #pragma unroll
        for (int t = 0; t < 4; ++t) {
            v16bf b = load_b_frag(wT, DIM, kk, col0 + t * 16, lane);
            acc[t] = wmma_bf16(a, b, acc[t]);
        }
    }

    const int half = lane >> 4;
    #pragma unroll
    for (int t = 0; t < 4; ++t) {
        int j = col0 + t * 16 + (lane & 15);          // 0..2303
        int which = j / DIM, rem = j % DIM;
        int h = rem >> 6, d = rem & 63;
        #pragma unroll
        for (int r = 0; r < 8; ++r) {
            int i = row0 + r + 8 * half;              // 0..8191
            int bidx = i >> 10, n = i & 1023;
            int bh = bidx * HEADS + h;
            unsigned short val = f2bf(acc[t][r]);
            if (which == 0)      qb[((size_t)bh * SEQ + n) * HDIM + d] = val;
            else if (which == 1) kb[((size_t)bh * SEQ + n) * HDIM + d] = val;
            else                 vT[((size_t)bh * HDIM + d) * SEQ + n] = val;
        }
    }
}

// ---------------- attention: scores + topk softmax + P@V, all in LDS ----------------
// grid(96, 16) block 256, dyn LDS = 64*1024*4 = 256 KB -> 1 WG/WGP, 2 waves/SIMD:
// B fragments are software double-buffered so loads run one iteration ahead of WMMA.
__global__ __launch_bounds__(256)
void attn_kernel(const unsigned short* __restrict__ qb,
                 const unsigned short* __restrict__ kbuf,
                 const unsigned short* __restrict__ vT,
                 unsigned short* __restrict__ aout) {
    extern __shared__ float S[];   // [64][1024]
    const int lane = threadIdx.x & 31, w = threadIdx.x >> 5;
    const int bh = blockIdx.x, rowblk = blockIdx.y;
    const unsigned short* qbh = qb   + (size_t)bh * SEQ * HDIM;
    const unsigned short* kbh = kbuf + (size_t)bh * SEQ * HDIM;
    const unsigned short* vbh = vT   + (size_t)bh * HDIM * SEQ;

    // ---- phase 1: S = (Q @ K^T) * scale  (wave: 16 rows x 512 cols) ----
    {
        const int rowband = w & 3, colhalf = w >> 2;
        const int row0 = rowblk * 64 + rowband * 16;
        v16bf a0 = load_a_frag(qbh, HDIM, row0, 0, lane);
        v16bf a1 = load_a_frag(qbh, HDIM, row0, 32, lane);
        const int mrow = rowband * 16 + 8 * (lane >> 4);
        int c0 = colhalf * 512;
        v16bf b0 = load_b_frag(kbh, HDIM, 0, c0, lane);
        v16bf b1 = load_b_frag(kbh, HDIM, 32, c0, lane);
        for (int t = 0; t < 32; ++t) {
            v16bf nb0 = b0, nb1 = b1;
            if (t < 31) {                       // issue next tile's loads first
                nb0 = load_b_frag(kbh, HDIM, 0, c0 + 16, lane);
                nb1 = load_b_frag(kbh, HDIM, 32, c0 + 16, lane);
            }
            if (t < 30) prefetch_b_frag(kbh, HDIM, 0, c0 + 32, lane);
            v8f c = (v8f){0,0,0,0,0,0,0,0};
            c = wmma_bf16(a0, b0, c);
            c = wmma_bf16(a1, b1, c);
            int col = c0 + (lane & 15);
            #pragma unroll
            for (int r = 0; r < 8; ++r) S[(mrow + r) * SEQ + col] = c[r] * SCALE;
            b0 = nb0; b1 = nb1; c0 += 16;
        }
    }
    __syncthreads();

    // ---- phase 2: per-row exact top-100 threshold + softmax (1 wave / row) ----
    for (int j = 0; j < 8; ++j) {
        const int row = w * 8 + j;
        float* Srow = S + row * SEQ;
        float sv[32]; unsigned int uk[32];
        float mx = -3.402823466e38f;
        #pragma unroll
        for (int i = 0; i < 32; ++i) {
            float s = Srow[lane * 32 + i];
            sv[i] = s;
            mx = fmaxf(mx, s);
            unsigned int u = __float_as_uint(s);
            uk[i] = (u & 0x80000000u) ? ~u : (u | 0x80000000u);  // monotonic key
        }
        for (int off = 16; off; off >>= 1) mx = fmaxf(mx, __shfl_xor(mx, off));
        // binary search: largest kv with count(key >= kv) >= TOPK
        unsigned int lo = 0u, hi = 0xFFFFFFFFu;
        while (lo < hi) {
            unsigned int mid = lo + ((hi - lo) >> 1) + 1u;
            int cnt = 0;
            #pragma unroll
            for (int i = 0; i < 32; ++i) cnt += (uk[i] >= mid) ? 1 : 0;
            for (int off = 16; off; off >>= 1) cnt += __shfl_xor(cnt, off);
            if (cnt >= TOPK) lo = mid; else hi = mid - 1u;
        }
        float sum = 0.0f;
        #pragma unroll
        for (int i = 0; i < 32; ++i) {
            float p = (uk[i] >= lo) ? __expf(sv[i] - mx) : 0.0f;
            sv[i] = p; sum += p;
        }
        for (int off = 16; off; off >>= 1) sum += __shfl_xor(sum, off);
        float rinv = 1.0f / sum;
        #pragma unroll
        for (int i = 0; i < 32; ++i) Srow[lane * 32 + i] = sv[i] * rinv;
    }
    __syncthreads();

    // ---- phase 3: O = P @ V  (wave: 16 rows x 32 d-cols, double-buffered B) ----
    {
        const int rowband = w & 3, n0base = (w >> 2) * 32;
        const int mrow0 = rowband * 16;
        const int m = lane & 15, half = lane >> 4;
        v8f o0 = (v8f){0,0,0,0,0,0,0,0};
        v8f o1 = (v8f){0,0,0,0,0,0,0,0};
        v16bf b0 = load_b_frag(vbh, SEQ, 0, n0base, lane);
        v16bf b1 = load_b_frag(vbh, SEQ, 0, n0base + 16, lane);
        for (int kk = 0; kk < SEQ; kk += 32) {
            v16bf nb0 = b0, nb1 = b1;
            if (kk + 32 < SEQ) {                // next iteration's V fragments
                nb0 = load_b_frag(vbh, SEQ, kk + 32, n0base, lane);
                nb1 = load_b_frag(vbh, SEQ, kk + 32, n0base + 16, lane);
            }
            // build A fragment from LDS probabilities (VALU overlaps the loads)
            const float* Sp = S + (mrow0 + m) * SEQ + kk + half * 8;
            v16us au = {};
            #pragma unroll
            for (int i = 0; i < 8; ++i) au[i] = f2bf(Sp[i]);
            #pragma unroll
            for (int i = 0; i < 8; ++i) au[8 + i] = f2bf(Sp[16 + i]);
            v16bf a = __builtin_bit_cast(v16bf, au);
            o0 = wmma_bf16(a, b0, o0);
            o1 = wmma_bf16(a, b1, o1);
            b0 = nb0; b1 = nb1;
        }
        const int b = bh / HEADS, h = bh % HEADS;
        const int nrow0 = rowblk * 64 + mrow0 + 8 * half;
        #pragma unroll
        for (int tt = 0; tt < 2; ++tt) {
            int d = n0base + tt * 16 + (lane & 15);
            v8f o = tt ? o1 : o0;
            #pragma unroll
            for (int r = 0; r < 8; ++r) {
                int nrow = nrow0 + r;
                aout[((size_t)b * SEQ + nrow) * DIM + h * HDIM + d] = f2bf(o[r]);
            }
        }
    }
}

// ---------------- output projection GEMM + bias ----------------
__global__ __launch_bounds__(256)
void proj_gemm_kernel(const unsigned short* __restrict__ ab,
                      const unsigned short* __restrict__ wT,
                      const float* __restrict__ bias,
                      float* __restrict__ out) {
    const int lane = threadIdx.x & 31, w = threadIdx.x >> 5;
    const int row0 = blockIdx.x * 64 + (w & 3) * 16;
    const int col0 = blockIdx.y * 128 + (w >> 2) * 64;

    v8f acc[4];
    #pragma unroll
    for (int t = 0; t < 4; ++t) acc[t] = (v8f){0,0,0,0,0,0,0,0};

    for (int kk = 0; kk < DIM; kk += 32) {
        v16bf a = load_a_frag(ab, DIM, row0, kk, lane);
        if (kk + 32 < DIM) {
            #pragma unroll
            for (int t = 0; t < 4; ++t)
                prefetch_b_frag(wT, DIM, kk + 32, col0 + t * 16, lane);
        }
        #pragma unroll
        for (int t = 0; t < 4; ++t) {
            v16bf b = load_b_frag(wT, DIM, kk, col0 + t * 16, lane);
            acc[t] = wmma_bf16(a, b, acc[t]);
        }
    }

    const int half = lane >> 4;
    #pragma unroll
    for (int t = 0; t < 4; ++t) {
        int j = col0 + t * 16 + (lane & 15);
        float bj = bias[j];
        #pragma unroll
        for (int r = 0; r < 8; ++r) {
            int i = row0 + r + 8 * half;
            out[(size_t)i * DIM + j] = acc[t][r] + bj;
        }
    }
}

// ---------------- host launcher ----------------
extern "C" void kernel_launch(void* const* d_in, const int* in_sizes, int n_in,
                              void* d_out, int out_size, void* d_ws, size_t ws_size,
                              hipStream_t stream) {
    const float* x      = (const float*)d_in[0];   // [8,1024,768]
    const float* w_qkv  = (const float*)d_in[1];   // [768,2304]
    const float* w_proj = (const float*)d_in[2];   // [768,768]
    const float* b_proj = (const float*)d_in[3];   // [768]
    float* out = (float*)d_out;                    // [8,1024,768] f32

    const size_t M = (size_t)BATCH * SEQ;          // 8192
    size_t off = 0;
    auto carve = [&](size_t bytes) {
        void* p = (char*)d_ws + off;
        off += (bytes + 255) & ~(size_t)255;
        return p;
    };
    unsigned short* xb     = (unsigned short*)carve(M * DIM * 2);
    unsigned short* wqkvT  = (unsigned short*)carve((size_t)3 * DIM * DIM * 2);
    unsigned short* wprojT = (unsigned short*)carve((size_t)DIM * DIM * 2);
    unsigned short* qb     = (unsigned short*)carve((size_t)BATCH * HEADS * SEQ * HDIM * 2);
    unsigned short* kb     = (unsigned short*)carve((size_t)BATCH * HEADS * SEQ * HDIM * 2);
    unsigned short* vT     = (unsigned short*)carve((size_t)BATCH * HEADS * HDIM * SEQ * 2);
    unsigned short* aout   = (unsigned short*)carve(M * DIM * 2);

    int nx = (int)(M * DIM);
    cast_bf16_kernel<<<(nx + 255) / 256, 256, 0, stream>>>(x, xb, nx);
    int nwq = DIM * 3 * DIM;
    transpose_bf16_kernel<<<(nwq + 255) / 256, 256, 0, stream>>>(w_qkv, wqkvT, DIM, 3 * DIM);
    int nwp = DIM * DIM;
    transpose_bf16_kernel<<<(nwp + 255) / 256, 256, 0, stream>>>(w_proj, wprojT, DIM, DIM);

    qkv_gemm_kernel<<<dim3(128, 18), 256, 0, stream>>>(xb, wqkvT, qb, kb, vT);

    attn_kernel<<<dim3(BATCH * HEADS, SEQ / 64), 256, 64 * SEQ * sizeof(float), stream>>>(
        qb, kb, vT, aout);

    proj_gemm_kernel<<<dim3(128, 6), 256, 0, stream>>>(aout, wprojT, b_proj, out);
}
